// TransformerBlock_44873818309053
// MI455X (gfx1250) — compile-verified, matrix operand order checked by reading, not run on hardware
//
#include <hip/hip_runtime.h>

typedef __attribute__((ext_vector_type(16))) __bf16 v16bf;
typedef __attribute__((ext_vector_type(8)))  float  v8f;

#define BB    4
#define SS    2048
#define DIM   1024
#define HID   4096
#define KSTEP 64

union BfFrag { v16bf v; unsigned int u[8]; };

static __device__ __forceinline__ unsigned short f2bf(float f) {
  unsigned int u = __float_as_uint(f);
  unsigned int r = u + 0x7FFFu + ((u >> 16) & 1u);   // round-to-nearest-even
  return (unsigned short)(r >> 16);
}

// CDNA5 async global->LDS copy, 16 bytes per lane (tracked by ASYNCcnt).
static __device__ __forceinline__ void async_load16(const unsigned short* g, unsigned lds_byte) {
  asm volatile("global_load_async_to_lds_b128 %0, %1, off"
               :: "v"(lds_byte), "v"(g)
               : "memory");
}
static __device__ __forceinline__ void wait_async_le12() {
  asm volatile("s_wait_asynccnt 0xc" ::: "memory");
}
static __device__ __forceinline__ void wait_async_0() {
  asm volatile("s_wait_asynccnt 0x0" ::: "memory");
}

// ------------- tiled transpose + fp32->bf16 convert: in[K,N] -> out[N,K] -------------
__global__ void __launch_bounds__(256) cvtT_kernel(const float* __restrict__ in,
                                                   unsigned short* __restrict__ out,
                                                   int Kd, int Nd) {
  __shared__ float t[32][33];
  const int k0 = blockIdx.y * 32, n0 = blockIdx.x * 32;
  const int tx = threadIdx.x, ty = threadIdx.y;   // (32, 8)
  #pragma unroll
  for (int i = 0; i < 4; ++i)
    t[ty * 4 + i][tx] = in[(size_t)(k0 + ty * 4 + i) * Nd + n0 + tx];
  __syncthreads();
  #pragma unroll
  for (int i = 0; i < 4; ++i)
    out[(size_t)(n0 + ty * 4 + i) * Kd + k0 + tx] = f2bf(t[tx][ty * 4 + i]);
}

// ------------- LayerNorm (f32 in) -> bf16 out, 1 block per row ----------------------
__global__ void __launch_bounds__(256) layernorm_kernel(const float* __restrict__ x,
                                                        const float* __restrict__ g,
                                                        const float* __restrict__ b,
                                                        unsigned short* __restrict__ out) {
  __shared__ float red[256];
  const int row = blockIdx.x;
  const int tid = threadIdx.x;
  const float* xp = x + (size_t)row * DIM;
  float4 v = *(const float4*)(xp + tid * 4);

  red[tid] = v.x + v.y + v.z + v.w; __syncthreads();
  #pragma unroll
  for (int st = 128; st > 0; st >>= 1) {
    if (tid < st) red[tid] += red[tid + st];
    __syncthreads();
  }
  const float mu = red[0] * (1.0f / DIM);
  __syncthreads();

  float dx = v.x - mu, dy = v.y - mu, dz = v.z - mu, dw = v.w - mu;
  red[tid] = dx * dx + dy * dy + dz * dz + dw * dw; __syncthreads();
  #pragma unroll
  for (int st = 128; st > 0; st >>= 1) {
    if (tid < st) red[tid] += red[tid + st];
    __syncthreads();
  }
  const float rstd = rsqrtf(red[0] * (1.0f / DIM) + 1e-5f);

  float4 gv = *(const float4*)(g + tid * 4);
  float4 bv = *(const float4*)(b + tid * 4);
  unsigned short o0 = f2bf(dx * rstd * gv.x + bv.x);
  unsigned short o1 = f2bf(dy * rstd * gv.y + bv.y);
  unsigned short o2 = f2bf(dz * rstd * gv.z + bv.z);
  unsigned short o3 = f2bf(dw * rstd * gv.w + bv.w);
  uint2 pk; pk.x = (unsigned int)o0 | ((unsigned int)o1 << 16);
  pk.y = (unsigned int)o2 | ((unsigned int)o3 << 16);
  *(uint2*)(out + (size_t)row * DIM + tid * 4) = pk;
}

// ------------- row softmax over SS cols: f32 in-place + bf16 copy -------------------
__global__ void __launch_bounds__(256) softmax_kernel(float* __restrict__ scores,
                                                      unsigned short* __restrict__ attn_bf) {
  __shared__ float red[256];
  const size_t row = blockIdx.x;
  const int tid = threadIdx.x;
  float* p = scores + row * (size_t)SS;
  float v[8];
  float mx = -3.402823466e38f;
  #pragma unroll
  for (int i = 0; i < 8; ++i) { v[i] = p[tid + i * 256]; mx = fmaxf(mx, v[i]); }
  red[tid] = mx; __syncthreads();
  #pragma unroll
  for (int st = 128; st > 0; st >>= 1) {
    if (tid < st) red[tid] = fmaxf(red[tid], red[tid + st]);
    __syncthreads();
  }
  mx = red[0]; __syncthreads();

  float s = 0.0f;
  #pragma unroll
  for (int i = 0; i < 8; ++i) { v[i] = __expf(v[i] - mx); s += v[i]; }
  red[tid] = s; __syncthreads();
  #pragma unroll
  for (int st = 128; st > 0; st >>= 1) {
    if (tid < st) red[tid] += red[tid + st];
    __syncthreads();
  }
  const float inv = 1.0f / red[0];
  unsigned short* q = attn_bf + row * (size_t)SS;
  #pragma unroll
  for (int i = 0; i < 8; ++i) {
    float r = v[i] * inv;
    p[tid + i * 256] = r;
    q[tid + i * 256] = f2bf(r);
  }
}

// ------------- bf16 WMMA GEMM, async double-buffered ---------------------------------
// C[M,N] = scale * (A[M,K] @ B^T) + bias (+residual) (+relu)
// A row-major [M,K] bf16; B row-major [N,K] bf16 (always "BT" layout).
// Block tile 128(M) x 256(N), 256 threads = 8 waves in 2x4 grid, wave tile 64x64,
// K stepped by 64.  16 WMMAs per 16 ds_load_b128 per 32-k sub-step.
// FLAGS bits: 1=bias, 2=residual, 4=relu, 8=outF, 16=outB, 32=outBT (batch-transposed)
template <int FLAGS>
__global__ void __launch_bounds__(256) gemm_kernel(
    const unsigned short* __restrict__ A, const unsigned short* __restrict__ B,
    const float* __restrict__ bias, const float* __restrict__ residual,
    float* __restrict__ outF, unsigned short* __restrict__ outB,
    unsigned short* __restrict__ outBT,
    int M, int N, int K, float scale, int btR) {
  constexpr bool HAS_BIAS = (FLAGS & 1)  != 0;
  constexpr bool HAS_RES  = (FLAGS & 2)  != 0;
  constexpr bool RELU     = (FLAGS & 4)  != 0;
  constexpr bool OUT_F    = (FLAGS & 8)  != 0;
  constexpr bool OUT_B    = (FLAGS & 16) != 0;
  constexpr bool OUT_BT   = (FLAGS & 32) != 0;

  __shared__ unsigned short ldsA[2][128 * KSTEP];   // [row][k]   16 KB each
  __shared__ unsigned short ldsB[2][256 * KSTEP];   // [col][k]   32 KB each

  const int tid   = threadIdx.x;
  const int lane  = tid & 31;
  const int wave  = tid >> 5;
  const int wm    = wave >> 2;   // 2 x 64-row strips
  const int wn    = wave & 3;    // 4 x 64-col strips
  const int lhalf = lane >> 4;
  const int lmod  = lane & 15;

  const int bm = blockIdx.y * 128;
  const int bn = blockIdx.x * 256;

  // 16-bit A/B operand layout (ISA 7.12.2): uint j holds K = base + (j&3)*2 (+1)
  int koff[8];
  #pragma unroll
  for (int j = 0; j < 8; ++j)
    koff[j] = ((j & 4) << 2) + (j & 3) * 2 + lhalf * 8;

  v8f acc[4][4] = {};

  // async tile loaders: per tile each thread moves 4x16B of A and 8x16B of B
  const int arow = tid >> 1;            // 0..127
  const int aseg = (tid & 1) * 32;      // 0 / 32 (elements)
  const unsigned short* aBase = A + (size_t)(bm + arow) * K + aseg;
  const unsigned short* bBase = B + (size_t)(bn + tid) * K;
  const unsigned ldsA0 = (unsigned)(size_t)&ldsA[0][0];
  const unsigned ldsA1 = (unsigned)(size_t)&ldsA[1][0];
  const unsigned ldsB0 = (unsigned)(size_t)&ldsB[0][0];
  const unsigned ldsB1 = (unsigned)(size_t)&ldsB[1][0];
  const unsigned loffA = (unsigned)(arow * KSTEP + aseg) * 2u;
  const unsigned loffB = (unsigned)(tid * KSTEP) * 2u;

  auto issue_tile = [&](int k0, int buf) {
    const unsigned short* ag = aBase + k0;
    const unsigned la = (buf ? ldsA1 : ldsA0) + loffA;
    #pragma unroll
    for (int c = 0; c < 4; ++c)
      async_load16(ag + c * 8, la + c * 16u);
    const unsigned short* bg = bBase + k0;
    const unsigned lb = (buf ? ldsB1 : ldsB0) + loffB;
    #pragma unroll
    for (int c = 0; c < 8; ++c)
      async_load16(bg + c * 8, lb + c * 16u);
  };

  const int T = K / KSTEP;
  issue_tile(0, 0);
  for (int t = 0; t < T; ++t) {
    if (t + 1 < T) {
      issue_tile((t + 1) * KSTEP, (t + 1) & 1);
      wait_async_le12();                // oldest 12 (tile t) have landed
    } else {
      wait_async_0();
    }
    __syncthreads();

    const unsigned short* bufA = &ldsA[t & 1][0];
    const unsigned short* bufB = &ldsB[t & 1][0];
    #pragma unroll
    for (int kb = 0; kb < KSTEP; kb += 32) {
      BfFrag afr[4];
      #pragma unroll
      for (int mt = 0; mt < 4; ++mt) {
        const unsigned short* ap = bufA + (wm * 64 + mt * 16 + lmod) * KSTEP + kb;
        #pragma unroll
        for (int j = 0; j < 8; ++j)
          afr[mt].u[j] = *(const unsigned int*)(ap + koff[j]);
      }
      #pragma unroll
      for (int nt = 0; nt < 4; ++nt) {
        BfFrag bfr;
        const unsigned short* bp = bufB + (wn * 64 + nt * 16 + lmod) * KSTEP + kb;
        #pragma unroll
        for (int j = 0; j < 8; ++j)
          bfr.u[j] = *(const unsigned int*)(bp + koff[j]);
        #pragma unroll
        for (int mt = 0; mt < 4; ++mt)
          acc[mt][nt] = __builtin_amdgcn_wmma_f32_16x16x32_bf16(
              false, afr[mt].v, false, bfr.v, (short)0, acc[mt][nt], false, false);
      }
    }
    __syncthreads();   // protect buf t&1 before it is re-targeted at t+2
  }

  // epilogue: C layout (ISA 7.12.2): lane%16 = N col, vgpr r = row (+8 for lanes 16..31)
  #pragma unroll
  for (int nt = 0; nt < 4; ++nt) {
    const int n = bn + wn * 64 + nt * 16 + lmod;
    const float bvl = HAS_BIAS ? bias[n] : 0.0f;
    #pragma unroll
    for (int mt = 0; mt < 4; ++mt) {
      #pragma unroll
      for (int r = 0; r < 8; ++r) {
        const int m = bm + wm * 64 + mt * 16 + lhalf * 8 + r;
        const size_t idx = (size_t)m * N + n;
        float c = acc[mt][nt][r] * scale + bvl;
        if (HAS_RES) c += residual[idx];
        if (RELU)    c = fmaxf(c, 0.0f);
        if (OUT_F)   outF[idx] = c;
        if (OUT_B)   outB[idx] = f2bf(c);
        if (OUT_BT) {                    // [batch][N][btR] transposed store (for V)
          const int b  = m / btR;
          const int mr = m - b * btR;
          outBT[(size_t)b * N * btR + (size_t)n * btR + mr] = f2bf(c);
        }
      }
    }
  }
}

// -------------------------------------------------------------------------------------
extern "C" void kernel_launch(void* const* d_in, const int* in_sizes, int n_in,
                              void* d_out, int out_size, void* d_ws, size_t ws_size,
                              hipStream_t stream) {
  (void)in_sizes; (void)n_in; (void)out_size; (void)ws_size;
  const float* x   = (const float*)d_in[0];
  const float* Wq  = (const float*)d_in[1];
  const float* bq  = (const float*)d_in[2];
  const float* Wk  = (const float*)d_in[3];
  const float* bk  = (const float*)d_in[4];
  const float* Wv  = (const float*)d_in[5];
  const float* bv  = (const float*)d_in[6];
  const float* Wo  = (const float*)d_in[7];
  const float* bo  = (const float*)d_in[8];
  const float* g1  = (const float*)d_in[9];
  const float* be1 = (const float*)d_in[10];
  const float* g2  = (const float*)d_in[11];
  const float* be2 = (const float*)d_in[12];
  const float* W1  = (const float*)d_in[13];
  const float* b1  = (const float*)d_in[14];
  const float* W2  = (const float*)d_in[15];
  const float* b2  = (const float*)d_in[16];

  float* outp  = (float*)d_out;
  float* attnp = outp + (size_t)BB * SS * DIM;     // attn returned as 2nd output

  unsigned short* wsp = (unsigned short*)d_ws;
  auto alloc = [&](size_t n) { unsigned short* p = wsp; wsp += n; return p; };
  unsigned short* WqT = alloc((size_t)DIM * DIM);
  unsigned short* WkT = alloc((size_t)DIM * DIM);
  unsigned short* WvT = alloc((size_t)DIM * DIM);
  unsigned short* WoT = alloc((size_t)DIM * DIM);
  unsigned short* W1T = alloc((size_t)HID * DIM);     // [HID, DIM]
  unsigned short* W2T = alloc((size_t)DIM * HID);     // [DIM, HID]
  unsigned short* xnb = alloc((size_t)BB * SS * DIM); // xn, later reused as xn2
  unsigned short* qb  = alloc((size_t)BB * SS * DIM); // q, later reused as ctx
  unsigned short* kb  = alloc((size_t)BB * SS * DIM);
  unsigned short* vT  = alloc((size_t)BB * DIM * SS); // v stored [b][D][S]
  unsigned short* ab  = alloc((size_t)BB * SS * SS);  // attn bf16
  unsigned short* hb  = alloc((size_t)BB * SS * HID); // MLP hidden bf16
  float* x2 = (float*)wsp;                            // post-attention residual (f32)

  const int M = BB * SS;                 // 8192 rows
  const float kScale = 1.0f / 32.0f;     // 1/sqrt(D)
  dim3 blk(256);
  dim3 tblk(32, 8);

  // transpose-convert all weights to bf16 [N,K]
  cvtT_kernel<<<dim3(DIM / 32, DIM / 32), tblk, 0, stream>>>(Wq, WqT, DIM, DIM);
  cvtT_kernel<<<dim3(DIM / 32, DIM / 32), tblk, 0, stream>>>(Wk, WkT, DIM, DIM);
  cvtT_kernel<<<dim3(DIM / 32, DIM / 32), tblk, 0, stream>>>(Wv, WvT, DIM, DIM);
  cvtT_kernel<<<dim3(DIM / 32, DIM / 32), tblk, 0, stream>>>(Wo, WoT, DIM, DIM);
  cvtT_kernel<<<dim3(HID / 32, DIM / 32), tblk, 0, stream>>>(W1, W1T, DIM, HID);
  cvtT_kernel<<<dim3(DIM / 32, HID / 32), tblk, 0, stream>>>(W2, W2T, HID, DIM);

  // xn = LN(x) -> bf16
  layernorm_kernel<<<M, blk, 0, stream>>>(x, g1, be1, xnb);

  // q/k = xn @ W + b -> bf16 row-major; v -> bf16 transposed per batch
  gemm_kernel<1 | 16><<<dim3(DIM / 256, M / 128), blk, 0, stream>>>(
      xnb, WqT, bq, nullptr, nullptr, qb, nullptr, M, DIM, DIM, 1.0f, 0);
  gemm_kernel<1 | 16><<<dim3(DIM / 256, M / 128), blk, 0, stream>>>(
      xnb, WkT, bk, nullptr, nullptr, kb, nullptr, M, DIM, DIM, 1.0f, 0);
  gemm_kernel<1 | 32><<<dim3(DIM / 256, M / 128), blk, 0, stream>>>(
      xnb, WvT, bv, nullptr, nullptr, nullptr, vT, M, DIM, DIM, 1.0f, SS);

  // scores[b] = scale * q[b] @ k[b]^T -> f32 straight into attn output region
  for (int b = 0; b < BB; ++b) {
    gemm_kernel<8><<<dim3(SS / 256, SS / 128), blk, 0, stream>>>(
        qb + (size_t)b * SS * DIM, kb + (size_t)b * SS * DIM, nullptr, nullptr,
        attnp + (size_t)b * SS * SS, nullptr, nullptr, SS, SS, DIM, kScale, 0);
  }

  // softmax in place (f32 output) + bf16 copy for attn@V
  softmax_kernel<<<BB * SS, blk, 0, stream>>>(attnp, ab);

  // ctx[b] = attn[b] @ v[b] -> bf16 (reuse q buffer); B = vT[b] is [DIM, SS]
  for (int b = 0; b < BB; ++b) {
    gemm_kernel<16><<<dim3(DIM / 256, SS / 128), blk, 0, stream>>>(
        ab + (size_t)b * SS * SS, vT + (size_t)b * DIM * SS, nullptr, nullptr,
        nullptr, qb + (size_t)b * SS * DIM, nullptr, SS, DIM, SS, 1.0f, 0);
  }

  // x2 = x + ctx @ Wo + bo  (f32)
  gemm_kernel<1 | 2 | 8><<<dim3(DIM / 256, M / 128), blk, 0, stream>>>(
      qb, WoT, bo, x, x2, nullptr, nullptr, M, DIM, DIM, 1.0f, 0);

  // xn2 = LN(x2) -> bf16 (reuse xn buffer)
  layernorm_kernel<<<M, blk, 0, stream>>>(x2, g2, be2, xnb);

  // h = relu(xn2 @ W1 + b1) -> bf16
  gemm_kernel<1 | 4 | 16><<<dim3(HID / 256, M / 128), blk, 0, stream>>>(
      xnb, W1T, b1, nullptr, nullptr, hb, nullptr, M, HID, DIM, 1.0f, 0);

  // out = x2 + h @ W2 + b2  (f32)
  gemm_kernel<1 | 2 | 8><<<dim3(DIM / 256, M / 128), blk, 0, stream>>>(
      hb, W2T, b2, x2, outp, nullptr, nullptr, M, DIM, HID, 1.0f, 0);
}
